// NoideAwareFeatureExtractor_8701603742383
// MI455X (gfx1250) — matrix operand order repaired and checked by reading, not compiled, not run on hardware
//
#include <hip/hip_runtime.h>
#include <hip/hip_bf16.h>
#include <math.h>

// ---------------------------------------------------------------------------
// NoiseAwareFeatureExtractor for MI455X (gfx1250, wave32, WMMA).
//
// FPS runs first; the heavy per-point MLP runs only on gathered rows.
// All GEMMs use v_wmma_f32_16x16x32_f16 (f16 operands, f32 accumulate).
// Weights are pre-swizzled into the WMMA B-operand lane layout; each k-step
// issues all 10 b128 fragment loads (A + 4xB in distinct registers) before
// the 4 WMMAs, so the scheduler can clause the loads and use partial
// s_wait_loadcnt values instead of a full stall per WMMA.
// ---------------------------------------------------------------------------

typedef __attribute__((ext_vector_type(16))) _Float16 v16h;
typedef __attribute__((ext_vector_type(8)))  _Float16 v8h;
typedef __attribute__((ext_vector_type(8)))  float    v8f;

#define BATCH 64
#define NPTS  4096
#define S1 500
#define S2 250
#define S3 125

union HU16 { v16h v; v8h h[2]; _Float16 e[16]; };

// ---------------- weight f32 -> f16, swizzled into WMMA B-fragment layout --
// dst layout: [kb][ntile][lane][h]  (contiguous 512 halves per (kb,ntile))
//   lane L: g = L>>4, c = L&15 ; half h -> K = kb*32 + g*16 + h, N = ntile*16+c
__global__ void k_convert_w_tiled(const float* __restrict__ src, _Float16* __restrict__ dst,
                                  int K, int N, int Kpad) {
  int t = blockIdx.x * blockDim.x + threadIdx.x;
  int total = Kpad * N;
  if (t >= total) return;
  int h    = t & 15;
  int lane = (t >> 4) & 31;
  int rest = t >> 9;                 // kb*ntilesN + ntile
  int ntilesN = N >> 4;
  int kb    = rest / ntilesN;
  int ntile = rest - kb * ntilesN;
  int k = kb * 32 + (lane >> 4) * 16 + h;
  int n = ntile * 16 + (lane & 15);
  float v = (k < K) ? src[(size_t)k * N + n] : 0.f;
  dst[t] = (_Float16)v;
}

// ---------------- per-point confidence MLP (3->64->1, sigmoid) -------------
__global__ void k_conf(const float* __restrict__ x, const float* __restrict__ cw1,
                       const float* __restrict__ cb1, const float* __restrict__ cw2,
                       const float* __restrict__ cb2, float* __restrict__ conf_ws,
                       float* __restrict__ conf_out) {
  int t = blockIdx.x * blockDim.x + threadIdx.x;
  if (t >= BATCH * NPTS) return;
  float x0 = x[(size_t)t * 3 + 0];
  float x1 = x[(size_t)t * 3 + 1];
  float x2 = x[(size_t)t * 3 + 2];
  float acc = cb2[0];
  #pragma unroll 8
  for (int j = 0; j < 64; j++) {
    float h = x0 * cw1[j] + x1 * cw1[64 + j] + x2 * cw1[128 + j] + cb1[j];
    h = fmaxf(h, 0.f);
    acc += h * cw2[j];
  }
  float c = 1.f / (1.f + __expf(-acc));
  conf_ws[t]  = c;
  conf_out[t] = c;
}

// ---------------- farthest point sampling (one WG per batch) ---------------
template <int PER>
__global__ void k_fps(const float* __restrict__ pts, int n, int npoint,
                      int* __restrict__ idx_out) {
  __shared__ float sval[256];
  __shared__ int   sidx[256];
  int b = blockIdx.x, tid = threadIdx.x;
  const float* p = pts + (size_t)b * n * 3;
  int* out = idx_out + (size_t)b * npoint;
  float px[PER], py[PER], pz[PER], dist[PER];
  #pragma unroll
  for (int t = 0; t < PER; t++) {
    int i = t * 256 + tid;
    dist[t] = 1e10f;
    if (i < n) { px[t] = p[i*3]; py[t] = p[i*3+1]; pz[t] = p[i*3+2]; }
    else       { px[t] = 0.f;    py[t] = 0.f;      pz[t] = 0.f;      }
  }
  int far = 0;
  for (int s = 0; s < npoint; s++) {
    if (tid == 0) out[s] = far;
    float cx = p[far*3], cy = p[far*3+1], cz = p[far*3+2];
    float bval = -1.f; int bidx = 0;
    #pragma unroll
    for (int t = 0; t < PER; t++) {
      int i = t * 256 + tid;
      if (i < n) {
        float dx = px[t]-cx, dy = py[t]-cy, dz = pz[t]-cz;
        float d = fminf(dist[t], dx*dx + dy*dy + dz*dz);
        dist[t] = d;
        if (d > bval) { bval = d; bidx = i; }
      }
    }
    sval[tid] = bval; sidx[tid] = bidx;
    __syncthreads();
    for (int o = 128; o > 0; o >>= 1) {
      if (tid < o) {
        float v2 = sval[tid+o]; int i2 = sidx[tid+o];
        if (v2 > sval[tid] || (v2 == sval[tid] && i2 < sidx[tid])) {
          sval[tid] = v2; sidx[tid] = i2;
        }
      }
      __syncthreads();
    }
    far = sidx[0];
    __syncthreads();
  }
}

// ---------------- gather xyz coords by index -------------------------------
__global__ void k_gather_xyz(const float* __restrict__ src, const int* __restrict__ idx,
                             float* __restrict__ dst, int nsrc, int S) {
  int t = blockIdx.x * blockDim.x + threadIdx.x;
  if (t >= BATCH * S) return;
  int b = t / S; int i = idx[t];
  const float* sp = src + ((size_t)b * nsrc + i) * 3;
  dst[(size_t)t*3 + 0] = sp[0];
  dst[(size_t)t*3 + 1] = sp[1];
  dst[(size_t)t*3 + 2] = sp[2];
}

// ---------------- build [conf,x] rows, K padded to 32 ----------------------
__global__ void k_build_fc(const float* __restrict__ x, const float* __restrict__ conf,
                           const int* __restrict__ idx, _Float16* __restrict__ dst) {
  int r = blockIdx.x * blockDim.x + threadIdx.x;
  if (r >= BATCH * S1) return;
  int b = r / S1; int i = idx[r];
  size_t pi = (size_t)b * NPTS + i;
  _Float16* d = dst + (size_t)r * 32;
  d[0] = (_Float16)conf[pi];
  d[1] = (_Float16)x[pi*3 + 0];
  d[2] = (_Float16)x[pi*3 + 1];
  d[3] = (_Float16)x[pi*3 + 2];
  for (int c = 4; c < 32; c++) d[c] = (_Float16)0.f;
}

// ---------------- build [xyz, feat] rows, K padded -------------------------
__global__ void k_build_concat(const float* __restrict__ xyz, const _Float16* __restrict__ feat,
                               _Float16* __restrict__ dst, int S, int C, int Kpad) {
  int r = blockIdx.x * blockDim.x + threadIdx.x;
  if (r >= BATCH * S) return;
  const float*    cp = xyz  + (size_t)r * 3;
  const _Float16* fp = feat + (size_t)r * C;
  _Float16*       d  = dst  + (size_t)r * Kpad;
  d[0] = (_Float16)cp[0]; d[1] = (_Float16)cp[1]; d[2] = (_Float16)cp[2];
  for (int c = 0; c < C; c++)        d[3 + c] = fp[c];
  for (int c = 3 + C; c < Kpad; c++) d[c] = (_Float16)0.f;
}

// ---------------- gather feature rows (vectorized 8 halves) ----------------
__global__ void k_gather_rows(const _Float16* __restrict__ src, const int* __restrict__ idx,
                              _Float16* __restrict__ dst, int nsrc, int S, int C) {
  int chunks = C >> 3;
  int t = blockIdx.x * blockDim.x + threadIdx.x;
  int total = BATCH * S * chunks;
  if (t >= total) return;
  int ch = t % chunks; int r = t / chunks;
  int b = r / S; int i = idx[r];
  const v8h* sp = (const v8h*)(src + ((size_t)b * nsrc + i) * C);
  v8h*       dp = (v8h*)(dst + (size_t)r * C);
  dp[ch] = sp[ch];
}

// ---------------- WMMA GEMM: Out = act(A@Wf + bias) ------------------------
// A: M x Kpad row-major f16 (Kpad%32==0). Wf: swizzled [kb][ntile][lane][16].
// Out: M x N f16. One wave computes a 16x64 strip: per k-step, all 10 b128
// fragment loads (A + 4xB, distinct registers) are issued before the 4 WMMAs
// so waits are partial. Requires N%64==0, M%16==0.
__global__ void __launch_bounds__(128, 1)
k_wmma_gemm(const _Float16* __restrict__ A, const _Float16* __restrict__ Wf,
            const float* __restrict__ bias, _Float16* __restrict__ Out,
            int M, int Kpad, int N, int do_relu) {
  int wave = (blockIdx.x * blockDim.x + threadIdx.x) >> 5;
  int lane = threadIdx.x & 31;
  int nstrips = N >> 6;                 // 64-wide strips
  int strips  = (M >> 4) * nstrips;
  if (wave >= strips) return;
  int mtile  = wave / nstrips;
  int nstrip = wave - mtile * nstrips;
  int g  = lane >> 4;
  int mr = lane & 15;
  int nkb     = Kpad >> 5;
  int ntilesN = N >> 4;

  const _Float16* arow  = A  + (size_t)(mtile * 16 + mr) * Kpad + g * 16;
  const _Float16* wlane = Wf + ((size_t)lane << 4) + (((size_t)nstrip << 2) << 9);
  size_t kstep = (size_t)ntilesN << 9;   // halves per kb step in Wf

  v8f acc[4];
  #pragma unroll
  for (int j = 0; j < 4; j++) acc[j] = (v8f){0.f,0.f,0.f,0.f,0.f,0.f,0.f,0.f};

  for (int kb = 0; kb < nkb; kb++) {
    // ---- issue all fragment loads for this k-step (distinct registers) ----
    HU16 a, b0, b1, b2, b3;
    const _Float16* ap = arow + kb * 32;
    a.h[0] = *(const v8h*)(ap);        // K = kb*32 + g*16 + 0..7
    a.h[1] = *(const v8h*)(ap + 8);    // K = kb*32 + g*16 + 8..15
    const _Float16* wp = wlane + (size_t)kb * kstep;
    b0.h[0] = *(const v8h*)(wp);
    b0.h[1] = *(const v8h*)(wp + 8);
    b1.h[0] = *(const v8h*)(wp + 512);
    b1.h[1] = *(const v8h*)(wp + 520);
    b2.h[0] = *(const v8h*)(wp + 1024);
    b2.h[1] = *(const v8h*)(wp + 1032);
    b3.h[0] = *(const v8h*)(wp + 1536);
    b3.h[1] = *(const v8h*)(wp + 1544);
    // ---- 4 WMMAs reuse the A fragment across the 64-wide strip ----
    acc[0] = __builtin_amdgcn_wmma_f32_16x16x32_f16(
        false, a.v, false, b0.v, (short)0, acc[0], false, false);
    acc[1] = __builtin_amdgcn_wmma_f32_16x16x32_f16(
        false, a.v, false, b1.v, (short)0, acc[1], false, false);
    acc[2] = __builtin_amdgcn_wmma_f32_16x16x32_f16(
        false, a.v, false, b2.v, (short)0, acc[2], false, false);
    acc[3] = __builtin_amdgcn_wmma_f32_16x16x32_f16(
        false, a.v, false, b3.v, (short)0, acc[3], false, false);
  }

  #pragma unroll
  for (int j = 0; j < 4; j++) {
    int col = ((nstrip << 2) + j) * 16 + mr;
    float bi = bias[col];
    #pragma unroll
    for (int v = 0; v < 8; v++) {
      float val = acc[j][v] + bi;
      if (do_relu) val = fmaxf(val, 0.f);
      int row = mtile * 16 + v + g * 8;   // C/D layout: VGPR v -> M = v + 8*g
      Out[(size_t)row * N + col] = (_Float16)val;
    }
  }
}

// ---------------- max over 125 rows -> (B,512) f32 -------------------------
__global__ void k_maxpool(const _Float16* __restrict__ l8, float* __restrict__ out) {
  int t = blockIdx.x * blockDim.x + threadIdx.x;
  if (t >= BATCH * 512) return;
  int b = t >> 9, c = t & 511;
  const _Float16* p = l8 + (size_t)b * S3 * 512 + c;
  float m = -1e30f;
  for (int r = 0; r < S3; r++) m = fmaxf(m, (float)p[(size_t)r * 512]);
  out[t] = m;
}

// ---------------------------------------------------------------------------
static inline int cdiv(int a, int b) { return (a + b - 1) / b; }

extern "C" void kernel_launch(void* const* d_in, const int* in_sizes, int n_in,
                              void* d_out, int out_size, void* d_ws, size_t ws_size,
                              hipStream_t stream) {
  const float* x    = (const float*)d_in[0];
  const float* cw1  = (const float*)d_in[1];
  const float* cb1  = (const float*)d_in[2];
  const float* cw2  = (const float*)d_in[3];
  const float* cb2  = (const float*)d_in[4];
  const float* w1   = (const float*)d_in[5];
  const float* b1   = (const float*)d_in[6];
  const float* w2   = (const float*)d_in[7];
  const float* b2   = (const float*)d_in[8];
  const float* w3   = (const float*)d_in[9];
  const float* b3   = (const float*)d_in[10];
  const float* w4   = (const float*)d_in[11];
  const float* b4   = (const float*)d_in[12];
  const float* w5   = (const float*)d_in[13];
  const float* b5   = (const float*)d_in[14];
  const float* w6   = (const float*)d_in[15];
  const float* b6   = (const float*)d_in[16];
  const float* pc1w = (const float*)d_in[17];
  const float* pc1b = (const float*)d_in[18];
  const float* pc2w = (const float*)d_in[19];
  const float* pc2b = (const float*)d_in[20];

  float* out_feat = (float*)d_out;            // (64,512)
  float* out_conf = out_feat + BATCH * 512;   // (64,4096,1)

  // ---- workspace carve-up (all offsets 256B aligned) ----
  char* ws = (char*)d_ws;
  size_t off = 0;
  auto alloc = [&](size_t bytes) -> char* {
    char* p = ws + off;
    off += (bytes + 255) & ~(size_t)255;
    return p;
  };
  float*    conf  = (float*)alloc((size_t)BATCH * NPTS * 4);
  int*      idx1  = (int*)alloc((size_t)BATCH * S1 * 4);
  int*      idx2  = (int*)alloc((size_t)BATCH * S2 * 4);
  int*      idx3  = (int*)alloc((size_t)BATCH * S3 * 4);
  float*    nx1   = (float*)alloc((size_t)BATCH * S1 * 3 * 4);
  float*    nx2   = (float*)alloc((size_t)BATCH * S2 * 3 * 4);
  _Float16* w1p   = (_Float16*)alloc((size_t)32  * 64  * 2);
  _Float16* w2p   = (_Float16*)alloc((size_t)64  * 256 * 2);
  _Float16* pc1wp = (_Float16*)alloc((size_t)256 * 256 * 2);
  _Float16* w3p   = (_Float16*)alloc((size_t)288 * 256 * 2);
  _Float16* w4p   = (_Float16*)alloc((size_t)256 * 384 * 2);
  _Float16* pc2wp = (_Float16*)alloc((size_t)384 * 384 * 2);
  _Float16* w5p   = (_Float16*)alloc((size_t)416 * 384 * 2);
  _Float16* w6p   = (_Float16*)alloc((size_t)384 * 512 * 2);
  _Float16* BUF0  = (_Float16*)alloc((size_t)BATCH * S1 * 416 * 2);  // 26.6 MB
  _Float16* BUF1  = (_Float16*)alloc((size_t)BATCH * S1 * 416 * 2);  // 26.6 MB

  const int M1 = BATCH * S1;   // 32000
  const int M2 = BATCH * S2;   // 16000
  const int M3 = BATCH * S3;   //  8000

  // ---- 1. weight conversion into WMMA-fragment layout ----
  #define CONVW(src, dst, K, N, KP) \
    k_convert_w_tiled<<<cdiv((KP)*(N), 256), 256, 0, stream>>>(src, dst, K, N, KP)
  CONVW(w1,   w1p,   4,   64,  32);
  CONVW(w2,   w2p,   64,  256, 64);
  CONVW(pc1w, pc1wp, 256, 256, 256);
  CONVW(w3,   w3p,   259, 256, 288);
  CONVW(w4,   w4p,   256, 384, 256);
  CONVW(pc2w, pc2wp, 384, 384, 384);
  CONVW(w5,   w5p,   387, 384, 416);
  CONVW(w6,   w6p,   384, 512, 384);
  #undef CONVW

  // ---- 2. confidence head over all points (also an output) ----
  k_conf<<<cdiv(BATCH * NPTS, 256), 256, 0, stream>>>(x, cw1, cb1, cw2, cb2,
                                                      conf, out_conf);

  // ---- 3. FPS level 1 + gathers ----
  k_fps<16><<<BATCH, 256, 0, stream>>>(x, NPTS, S1, idx1);
  k_gather_xyz<<<cdiv(M1, 256), 256, 0, stream>>>(x, idx1, nx1, NPTS, S1);
  k_build_fc<<<cdiv(M1, 256), 256, 0, stream>>>(x, conf, idx1, BUF0);

  // grid: one wave per 16x64 strip, 4 waves (128 thr) per block
  #define GEMM(A, W, Bv, O, M, KP, N, RELU) \
    k_wmma_gemm<<<cdiv(((M) >> 4) * ((N) >> 6), 4), 128, 0, stream>>>(A, W, Bv, O, M, KP, N, RELU)

  // ---- 4. per-point MLP on gathered rows + SA1 ----
  GEMM(BUF0, w1p,   b1,   BUF1, M1, 32,  64,  1);  // act64
  GEMM(BUF1, w2p,   b2,   BUF0, M1, 64,  256, 1);  // nf1
  GEMM(BUF0, pc1wp, pc1b, BUF1, M1, 256, 256, 0);  // l3
  k_build_concat<<<cdiv(M1, 256), 256, 0, stream>>>(nx1, BUF1, BUF0, S1, 256, 288);
  GEMM(BUF0, w3p,   b3,   BUF1, M1, 288, 256, 1);  // l4
  GEMM(BUF1, w4p,   b4,   BUF0, M1, 256, 384, 1);  // l5

  // ---- 5. FPS level 2 + SA2 ----
  k_fps<2><<<BATCH, 256, 0, stream>>>(nx1, S1, S2, idx2);
  k_gather_xyz<<<cdiv(M2, 256), 256, 0, stream>>>(nx1, idx2, nx2, S1, S2);
  k_gather_rows<<<cdiv(M2 * (384 / 8), 256), 256, 0, stream>>>(BUF0, idx2, BUF1, S1, S2, 384); // nf2
  GEMM(BUF1, pc2wp, pc2b, BUF0, M2, 384, 384, 0);  // l6
  k_build_concat<<<cdiv(M2, 256), 256, 0, stream>>>(nx2, BUF0, BUF1, S2, 384, 416);
  GEMM(BUF1, w5p,   b5,   BUF0, M2, 416, 384, 1);  // l7

  // ---- 6. FPS level 3 + final layer + max pool ----
  k_fps<1><<<BATCH, 256, 0, stream>>>(nx2, S2, S3, idx3);
  k_gather_rows<<<cdiv(M3 * (384 / 8), 256), 256, 0, stream>>>(BUF0, idx3, BUF1, S2, S3, 384); // nf3
  GEMM(BUF1, w6p,   b6,   BUF0, M3, 384, 512, 1);  // l8
  k_maxpool<<<cdiv(BATCH * 512, 256), 256, 0, stream>>>(BUF0, out_feat);

  #undef GEMM
  (void)in_sizes; (void)n_in; (void)out_size; (void)ws_size;
}
